// W8A16LinearStatic_20469814133415
// MI455X (gfx1250) — matrix operand order repaired and checked by reading, not compiled, not run on hardware
//
#include <hip/hip_runtime.h>

typedef __attribute__((ext_vector_type(16))) _Float16 v16h;
typedef __attribute__((ext_vector_type(8)))  _Float16 v8h;
typedef __attribute__((ext_vector_type(8)))  float    v8f;
typedef __attribute__((ext_vector_type(4)))  float    v4f;
typedef __attribute__((ext_vector_type(4)))  int      v4i;

#define M_TOT 8192   // B * S = 4 * 2048
#define N_TOT 4096   // DOUT
#define K_TOT 4096   // DIN
#define BM 128
#define BN 256
#define BK 32
#define KSTEPS (K_TOT / BK)
#define LDS_LD 40    // 32 halves + 8 pad (80B row, keeps 16B alignment, spreads banks)

struct H16Pair { v8h lo, hi; };

__device__ __forceinline__ v16h make_v16h(v8h lo, v8h hi) {
  H16Pair p{lo, hi};
  return __builtin_bit_cast(v16h, p);
}

__global__ __launch_bounds__(256)
void w8a16_wmma_f16_kernel(const float* __restrict__ x,
                           const int*   __restrict__ w,
                           const float* __restrict__ scale,
                           const int*   __restrict__ offset,
                           float*       __restrict__ out)
{
  __shared__ alignas(16) _Float16 sA[2][BM][LDS_LD];   // 20 KB
  __shared__ alignas(16) _Float16 sB[2][BN][LDS_LD];   // 40 KB

  const int t  = threadIdx.x;
  const int bm = blockIdx.y * BM;
  const int bn = blockIdx.x * BN;

  // ---- activation staging: each thread owns half a 32-wide row (16 f32)
  const int arow = t >> 1;           // 0..127
  const int acol = (t & 1) << 4;     // 0 or 16
  const float* xg = x + (size_t)(bm + arow) * K_TOT + acol;

  // ---- weight staging: each thread owns one full 32-wide row (32 i32)
  const int wrow = t;                // 0..255
  const int*   wg  = w + (size_t)(bn + wrow) * K_TOT;
  const float  wsc = scale[bn + wrow];          // per-output-channel, loop invariant
  const float  wof = (float)offset[bn + wrow];

  // ---- wave / lane decomposition: 8 waves in 2 (M) x 4 (N); each wave: 64x64 of C
  const int wave = t >> 5;
  const int lane = t & 31;
  const int wm   = wave >> 2;        // 0..1 -> 64 rows
  const int wn   = wave & 3;         // 0..3 -> 64 cols
  const int lh   = lane & 15;
  const int lhi  = lane >> 4;

  v8f acc[4][4];
#pragma unroll
  for (int mt = 0; mt < 4; ++mt)
#pragma unroll
    for (int nt = 0; nt < 4; ++nt)
      acc[mt][nt] = v8f{0.f, 0.f, 0.f, 0.f, 0.f, 0.f, 0.f, 0.f};

  v4f ax[4];   // staged activations (16 f32)
  v4i aw[8];   // staged int8-in-int32 weights (32)

  auto gload = [&](int kk) {
    const float* xp = xg + kk;
    const int*   wp = wg + kk;
#pragma unroll
    for (int i = 0; i < 4; ++i) ax[i] = *(const v4f*)(xp + 4 * i);
#pragma unroll
    for (int i = 0; i < 8; ++i) aw[i] = *(const v4i*)(wp + 4 * i);
  };

  auto sstore = [&](int s) {
    _Float16* pa = &sA[s][arow][acol];
    _Float16* pb = &sB[s][wrow][0];
    v8h ha[2], hb[4];
#pragma unroll
    for (int i = 0; i < 8; ++i) {
      ha[0][i] = (_Float16)ax[i >> 2][i & 3];
      ha[1][i] = (_Float16)ax[2 + (i >> 2)][i & 3];
    }
#pragma unroll
    for (int v = 0; v < 4; ++v)
#pragma unroll
      for (int i = 0; i < 8; ++i)
        hb[v][i] = (_Float16)(((float)aw[2 * v + (i >> 2)][i & 3] - wof) * wsc);
    *(v8h*)(pa)      = ha[0];
    *(v8h*)(pa + 8)  = ha[1];
#pragma unroll
    for (int v = 0; v < 4; ++v) *(v8h*)(pb + 8 * v) = hb[v];
  };

  auto compute = [&](int s) {
    // Load ALL fragments first (dedicated registers) so the 16 WMMAs issue
    // back-to-back with a single batched ds-wait instead of 4 drained waits.
    v16h bfrag[4], afrag[4];
#pragma unroll
    for (int nt = 0; nt < 4; ++nt) {
      const int col = wn * 64 + nt * 16 + lh;
      v8h blo = *(const v8h*)&sB[s][col][lhi * 8];
      v8h bhi = *(const v8h*)&sB[s][col][16 + lhi * 8];
      bfrag[nt] = make_v16h(blo, bhi);
    }
#pragma unroll
    for (int mt = 0; mt < 4; ++mt) {
      const int row = wm * 64 + mt * 16 + lh;
      v8h alo = *(const v8h*)&sA[s][row][lhi * 8];
      v8h ahi = *(const v8h*)&sA[s][row][16 + lhi * 8];
      afrag[mt] = make_v16h(alo, ahi);
    }
#pragma unroll
    for (int mt = 0; mt < 4; ++mt)
#pragma unroll
      for (int nt = 0; nt < 4; ++nt)
        acc[mt][nt] = __builtin_amdgcn_wmma_f32_16x16x32_f16(
            false, afrag[mt], false, bfrag[nt], (short)0, acc[mt][nt], false, false);
  };

  // ---- software-pipelined main loop (double-buffered LDS)
  gload(0);
  sstore(0);
  int s = 0;
#pragma unroll 1
  for (int kt = 0; kt < KSTEPS; ++kt) {
    __syncthreads();
    if (kt + 1 < KSTEPS) gload((kt + 1) * BK);
    compute(s);
    __syncthreads();
    if (kt + 1 < KSTEPS) sstore(s ^ 1);
    s ^= 1;
  }

  // ---- epilogue: C/D layout -> VGPR i holds row (lhi*8 + i), col = lh
#pragma unroll
  for (int mt = 0; mt < 4; ++mt) {
#pragma unroll
    for (int nt = 0; nt < 4; ++nt) {
      const int gcol = bn + wn * 64 + nt * 16 + lh;
#pragma unroll
      for (int i = 0; i < 8; ++i) {
        const int grow = bm + wm * 64 + mt * 16 + lhi * 8 + i;
        out[(size_t)grow * N_TOT + gcol] = acc[mt][nt][i];
      }
    }
  }
}

extern "C" void kernel_launch(void* const* d_in, const int* in_sizes, int n_in,
                              void* d_out, int out_size, void* d_ws, size_t ws_size,
                              hipStream_t stream) {
  (void)in_sizes; (void)n_in; (void)out_size; (void)d_ws; (void)ws_size;
  const float* x      = (const float*)d_in[0];
  const int*   weight = (const int*)d_in[1];
  const float* wscale = (const float*)d_in[2];
  const int*   woffs  = (const int*)d_in[3];
  float*       out    = (float*)d_out;

  dim3 grid(N_TOT / BN, M_TOT / BM);   // (16, 64)
  w8a16_wmma_f16_kernel<<<grid, 256, 0, stream>>>(x, weight, wscale, woffs, out);
}
